// SpeechAttDecVol3_58884001628652
// MI455X (gfx1250) — matrix operand order repaired and checked by reading, not compiled, not run on hardware
//
#include <hip/hip_runtime.h>
#include <hip/hip_bf16.h>
#include <math.h>

typedef __bf16 bf16_t;
typedef __attribute__((ext_vector_type(16))) __bf16 v16bf;
typedef __attribute__((ext_vector_type(8)))  float  v8f;

#define LSEQ 41
#define NB   4096
#define HD   256
#define G3   768      // 3*H
#define G6   1536     // 6*H
#define TQ   40
#define OV   27
#define OQ   10

// ---------------------------------------------------------------------------
// Setup kernels (run once, all invariant across the 40 recurrent steps)
// ---------------------------------------------------------------------------

// wall[c,k] (bf16, 1536x256): rows 0..767 = Wih @ Wc[:, :H]; rows 768..1535 = Whh
__global__ __launch_bounds__(256) void k_build_wall(const float* __restrict__ Wc,
                                                    const float* __restrict__ Wih,
                                                    const float* __restrict__ Whh,
                                                    bf16_t* __restrict__ wall) {
  int idx = blockIdx.x * 256 + threadIdx.x;          // over 1536*256
  int c = idx >> 8, k = idx & 255;
  float s;
  if (c < G3) {
    s = 0.f;
    const float* wr = Wih + (size_t)c * HD;
    for (int p = 0; p < HD; ++p) s += wr[p] * Wc[(size_t)p * (2 * HD) + k];
  } else {
    s = Whh[(size_t)(c - G3) * HD + k];
  }
  wall[idx] = (bf16_t)s;
}

// ctx[n,k] = mean over l of enc[l,n,k]  (uniform attention -> time-invariant)
__global__ __launch_bounds__(256) void k_ctx_mean(const float* __restrict__ enc,
                                                  float* __restrict__ ctx) {
  int idx = blockIdx.x * 256 + threadIdx.x;          // over N*H
  float s = 0.f;
  for (int l = 0; l < LSEQ; ++l) s += enc[(size_t)l * NB * HD + idx];
  ctx[idx] = s * (1.0f / (float)LSEQ);
}

// cpart[n,i] = bc[i] + sum_k ctx[n,k] * Wc[i, H+k]
__global__ __launch_bounds__(256) void k_cpart(const float* __restrict__ ctx,
                                               const float* __restrict__ Wc,
                                               const float* __restrict__ bc,
                                               float* __restrict__ cpart) {
  int idx = blockIdx.x * 256 + threadIdx.x;          // over N*H
  int n = idx >> 8, i = idx & 255;
  const float* cr = ctx + (size_t)n * HD;
  const float* wr = Wc + (size_t)i * (2 * HD) + HD;
  float s = bc[i];
  for (int k = 0; k < HD; ++k) s += cr[k] * wr[k];
  cpart[idx] = s;
}

// ci[n,c] = bih[c] + sum_i cpart[n,i] * Wih[c,i]
__global__ __launch_bounds__(256) void k_ci(const float* __restrict__ cpart,
                                            const float* __restrict__ Wih,
                                            const float* __restrict__ bih,
                                            float* __restrict__ ci) {
  int idx = blockIdx.x * 256 + threadIdx.x;          // over N*768
  int n = idx / G3, c = idx % G3;
  const float* cp = cpart + (size_t)n * HD;
  const float* wr = Wih + (size_t)c * HD;
  float s = bih[c];
  for (int i = 0; i < HD; ++i) s += cp[i] * wr[i];
  ci[idx] = s;
}

__global__ __launch_bounds__(256) void k_hinit(const float* __restrict__ hidden,
                                               float* __restrict__ h,
                                               bf16_t* __restrict__ hbf) {
  int idx = blockIdx.x * 256 + threadIdx.x;          // over N*H
  float v = hidden[idx];
  h[idx] = v;
  hbf[idx] = (bf16_t)v;
}

// ---------------------------------------------------------------------------
// Fully fused GRU step (one launch per timestep, no intermediate G buffer):
//   G = hbf(16 rows x 256) @ wall(1536 x 256)^T  via bf16 WMMA, then gates.
// Block = 8 waves = one 16-row M-tile of the batch.
// Wave w owns column-tiles {2w, 2w+1} + 16k (k=0..5): 12 accumulators, which
// is exactly the 6-tile gate sextuple (i, i+256, ..., i+1280) for 32 hidden
// columns -> the whole GRU gate update happens in registers per (lane, vgpr).
// ---------------------------------------------------------------------------
__global__ __launch_bounds__(256) void k_step_fused(const bf16_t* __restrict__ wall,
                                                    const float* __restrict__ ci,
                                                    const float* __restrict__ bhh,
                                                    float* __restrict__ h,
                                                    bf16_t* __restrict__ hbf,
                                                    bf16_t* __restrict__ outs_t) {
  const int wave = threadIdx.x >> 5;
  const int lane = threadIdx.x & 31;
  const int lhi  = lane >> 4;                        // 0/1
  const int lm   = lane & 15;
  const int mBase = blockIdx.x * 16;                 // 256 blocks

  union AB { v16bf v; float4 f4[2]; };
  v8f acc[12] = {};                                  // [bl*6 + kk]

  // A fragment base: row mBase+lm, K-offset lhi*8 (16-bit 16x32 layout)
  const bf16_t* aBase = hbf + (size_t)(mBase + lm) * HD + lhi * 8;
  // B fragment base: wall row (2w)*16 + lm, K-offset lhi*8 (32x16 layout)
  const bf16_t* bBase = wall + (size_t)(2 * wave * 16 + lm) * HD + lhi * 8;

#pragma unroll
  for (int k = 0; k < 8; ++k) {                      // K-step of 32
    AB a;
    a.f4[0] = *(const float4*)(aBase + k * 32);
    a.f4[1] = *(const float4*)(aBase + k * 32 + 16);
#pragma unroll
    for (int kk = 0; kk < 6; ++kk) {
#pragma unroll
      for (int bl = 0; bl < 2; ++bl) {
        const bf16_t* bp = bBase + k * 32 + (size_t)(bl * 16 + kk * 256) * HD;
        AB b;
        b.f4[0] = *(const float4*)(bp);
        b.f4[1] = *(const float4*)(bp + 16);
        acc[bl * 6 + kk] = __builtin_amdgcn_wmma_f32_16x16x32_bf16(
            false, a.v, false, b.v, (short)0, acc[bl * 6 + kk], false, false);
      }
    }
  }

  // All waves have finished reading this block's hbf rows before anyone
  // overwrites them below.
  __syncthreads();

#pragma unroll
  for (int bl = 0; bl < 2; ++bl) {
    const int i = wave * 32 + bl * 16 + lm;          // hidden column 0..255
    const float b_r = bhh[i];
    const float b_z = bhh[i + 256];
    const float b_n = bhh[i + 512];
#pragma unroll
    for (int v = 0; v < 8; ++v) {
      const int n = mBase + lhi * 8 + v;             // batch row (C/D layout)
      const float* cn = ci + (size_t)n * G3;
      float ir  = acc[bl * 6 + 0][v] + cn[i];
      float iz  = acc[bl * 6 + 1][v] + cn[i + 256];
      float inn = acc[bl * 6 + 2][v] + cn[i + 512];
      float hr  = acc[bl * 6 + 3][v] + b_r;
      float hz  = acc[bl * 6 + 4][v] + b_z;
      float hn  = acc[bl * 6 + 5][v] + b_n;
      float r = 1.f / (1.f + expf(-(ir + hr)));
      float z = 1.f / (1.f + expf(-(iz + hz)));
      float g = tanhf(inn + r * hn);
      const size_t idx = (size_t)n * HD + i;
      float hnew = (1.f - z) * g + z * h[idx];
      h[idx] = hnew;
      bf16_t hb = (bf16_t)hnew;
      hbf[idx] = hb;
      outs_t[idx] = hb;
    }
  }
}

// ---------------------------------------------------------------------------
// Epilogue 1: yt[n,c,t] = outs[t,n,:] . Wo[c,:] + bo[c]   (layout (N,27,40))
// ---------------------------------------------------------------------------
__global__ __launch_bounds__(256) void k_proj(const bf16_t* __restrict__ outs,
                                              const float* __restrict__ Wo,
                                              const float* __restrict__ bo,
                                              float* __restrict__ yt) {
  __shared__ float rows[8][HD];
  const int t = blockIdx.x;
  const int nBase = blockIdx.y * 8;
  const bf16_t* src = outs + ((size_t)t * NB + nBase) * HD;
  for (int r = 0; r < 8; ++r)
    rows[r][threadIdx.x] = (float)src[(size_t)r * HD + threadIdx.x];
  __syncthreads();
  int tid = threadIdx.x;
  if (tid < 8 * OV) {
    int c = tid % OV, r = tid / OV;
    const float* w = Wo + (size_t)c * HD;
    float s = bo[c];
    for (int k = 0; k < HD; ++k) s += rows[r][k] * w[k];
    yt[((size_t)(nBase + r) * OV + c) * TQ + t] = s;
  }
}

// ---------------------------------------------------------------------------
// Epilogue 2: q[n,j,c] = yt[n,c,:] . Wq[j,:] + bq[j], log_softmax over c
// ---------------------------------------------------------------------------
__global__ __launch_bounds__(256) void k_quant_lsm(const float* __restrict__ yt,
                                                   const float* __restrict__ Wq,
                                                   const float* __restrict__ bq,
                                                   float* __restrict__ out) {
  const int wave = threadIdx.x >> 5;
  const int lane = threadIdx.x & 31;
  const int job = blockIdx.x * 8 + wave;             // N*10 jobs
  const int n = job / OQ, j = job % OQ;
  float val = -1e30f;
  if (lane < OV) {
    const float* y = yt + ((size_t)n * OV + lane) * TQ;
    const float* w = Wq + (size_t)j * TQ;
    float s = bq[j];
    for (int t = 0; t < TQ; ++t) s += y[t] * w[t];
    val = s;
  }
  float m = val;
  for (int off = 16; off; off >>= 1) m = fmaxf(m, __shfl_xor(m, off, 32));
  float e = (lane < OV) ? expf(val - m) : 0.f;
  for (int off = 16; off; off >>= 1) e += __shfl_xor(e, off, 32);
  float lse = logf(e);
  if (lane < OV) out[((size_t)n * OQ + j) * OV + lane] = val - m - lse;
}

// ---------------------------------------------------------------------------
extern "C" void kernel_launch(void* const* d_in, const int* in_sizes, int n_in,
                              void* d_out, int out_size, void* d_ws, size_t ws_size,
                              hipStream_t stream) {
  const float* enc    = (const float*)d_in[0];   // (41, 4096, 256)
  const float* hidden = (const float*)d_in[1];   // (1, 4096, 256)
  const float* Wc     = (const float*)d_in[2];   // (256, 512)
  const float* bc     = (const float*)d_in[3];   // (256)
  const float* Wih    = (const float*)d_in[4];   // (768, 256)
  const float* Whh    = (const float*)d_in[5];   // (768, 256)
  const float* bih    = (const float*)d_in[6];   // (768)
  const float* bhh    = (const float*)d_in[7];   // (768)
  const float* Wo     = (const float*)d_in[8];   // (27, 256)
  const float* bo     = (const float*)d_in[9];   // (27)
  const float* Wq     = (const float*)d_in[10];  // (10, 40)
  const float* bq     = (const float*)d_in[11];  // (10)
  float* out = (float*)d_out;                    // (4096, 10, 27)

  char* p = (char*)d_ws;
  auto carve = [&](size_t bytes) { char* r = p; p += (bytes + 255) & ~(size_t)255; return (void*)r; };
  bf16_t* wall  = (bf16_t*)carve((size_t)G6 * HD * 2);
  float*  ci    = (float*) carve((size_t)NB * G3 * 4);
  float*  ctx   = (float*) carve((size_t)NB * HD * 4);
  float*  cpart = (float*) carve((size_t)NB * HD * 4);
  float*  hbuf  = (float*) carve((size_t)NB * HD * 4);
  bf16_t* hbf   = (bf16_t*)carve((size_t)NB * HD * 2);
  bf16_t* outs  = (bf16_t*)carve((size_t)TQ * NB * HD * 2);
  float*  yt    = (float*) carve((size_t)NB * OV * TQ * 4);

  // one-time precompute (attention context is uniform-mean -> time-invariant)
  k_build_wall<<<(G6 * HD) / 256, 256, 0, stream>>>(Wc, Wih, Whh, wall);
  k_ctx_mean<<<(NB * HD) / 256, 256, 0, stream>>>(enc, ctx);
  k_cpart<<<(NB * HD) / 256, 256, 0, stream>>>(ctx, Wc, bc, cpart);
  k_ci<<<(NB * G3) / 256, 256, 0, stream>>>(cpart, Wih, bih, ci);
  k_hinit<<<(NB * HD) / 256, 256, 0, stream>>>(hidden, hbuf, hbf);

  // 40 sequential, fully fused GRU steps (WMMA GEMM + gates in one kernel)
  for (int t = 0; t < TQ; ++t) {
    k_step_fused<<<NB / 16, 256, 0, stream>>>(wall, ci, bhh, hbuf, hbf,
                                              outs + (size_t)t * NB * HD);
  }

  // output projection + time quantization + log_softmax
  k_proj<<<dim3(TQ, NB / 8), 256, 0, stream>>>(outs, Wo, bo, yt);
  k_quant_lsm<<<(NB * OQ) / 8, 256, 0, stream>>>(yt, Wq, bq, out);
}
